// SpikingHiddenNetwork_70866960384643
// MI455X (gfx1250) — matrix operand order repaired and checked by reading, not compiled, not run on hardware
//
#include <hip/hip_runtime.h>

typedef _Float16 v16h __attribute__((ext_vector_type(16)));
typedef _Float16 v8h  __attribute__((ext_vector_type(8)));
typedef float    v8f  __attribute__((ext_vector_type(8)));
typedef float    v4f  __attribute__((ext_vector_type(4)));

#define B_TOTAL 4096
#define NIN     784
#define NHID    1024
#define NOUT    256
#define NSTEP   32
#define K1P     800      // layer-1 K padded to multiple of 32
#define BM      64       // batch rows per workgroup (4 row-tiles)
#define NTHREADS 256     // 8 waves -> 2 waves/SIMD, ~512-VGPR budget per wave

#define XI_PITCH 816     // halves per xi row (16B-aligned row stride)
#define S1_PITCH 1040    // halves per s1 row

#define DECAY1 0.8187307530779818f   // exp(-1/5)
#define DECAY2 0.8187307530779818f

static __device__ __forceinline__ v16h cat16(v8h lo, v8h hi) {
  v16h r;
#pragma unroll
  for (int i = 0; i < 8; ++i) { r[i] = lo[i]; r[i + 8] = hi[i]; }
  return r;
}

// ---- weight conversion: f32 -> f16, W1 padded K 784->800 --------------------
__global__ void cvt_w1(const float* __restrict__ W1, _Float16* __restrict__ W1h) {
  int idx = blockIdx.x * 256 + threadIdx.x;          // over NHID*K1P
  if (idx >= NHID * K1P) return;
  int n = idx / K1P, k = idx - n * K1P;
  W1h[idx] = (k < NIN) ? (_Float16)W1[n * NIN + k] : (_Float16)0.0f;
}

__global__ void cvt_w2(const float* __restrict__ W2, _Float16* __restrict__ W2h) {
  int idx = blockIdx.x * 256 + threadIdx.x;          // over NOUT*NHID
  if (idx >= NOUT * NHID) return;
  W2h[idx] = (_Float16)W2[idx];
}

// ---- fused 32-step spiking network, one 64-row batch tile per workgroup -----
__global__ void __launch_bounds__(NTHREADS, 1)
snn_kernel(const float* __restrict__ x, const float* __restrict__ b1,
           const _Float16* __restrict__ W1h, const _Float16* __restrict__ W2h,
           const float* __restrict__ noise, float* __restrict__ out) {
  extern __shared__ char smem[];
  _Float16* xi = (_Float16*)smem;                         // [BM][XI_PITCH]
  _Float16* s1 = (_Float16*)(smem + BM * XI_PITCH * 2);   // [BM][S1_PITCH]

  const int tid  = threadIdx.x;
  const int lane = tid & 31;
  const int wv   = tid >> 5;        // wave 0..7
  const int ln16 = lane & 15;
  const int hi   = lane >> 4;       // K-half selector for WMMA operands
  const int r0   = blockIdx.x * BM;

  // zero xi pad columns once (cols 784..815 never rewritten)
  for (int i = tid; i < BM * (XI_PITCH - NIN); i += NTHREADS) {
    int r = i / (XI_PITCH - NIN), c = NIN + i % (XI_PITCH - NIN);
    xi[r * XI_PITCH + c] = (_Float16)0.0f;
  }

  // persistent per-lane state (WMMA C/D layout). v1acc/v2acc always hold
  // decay*v*(1-s), i.e. they are the ready-to-accumulate WMMA C operand.
  v8f v1acc[4][8];                 // [row-tile][col-tile] layer-1 membrane
  v8f v2acc[4][2];                 // layer-2 membrane
  v8f nsp[4][2];                   // spike counts (output)
#pragma unroll
  for (int rt = 0; rt < 4; ++rt) {
#pragma unroll
    for (int nt = 0; nt < 8; ++nt)
#pragma unroll
      for (int j = 0; j < 8; ++j) v1acc[rt][nt][j] = 0.0f;
#pragma unroll
    for (int nt = 0; nt < 2; ++nt)
#pragma unroll
      for (int j = 0; j < 8; ++j) { v2acc[rt][nt][j] = 0.0f; nsp[rt][nt][j] = 0.0f; }
  }
  float b1v[8];
#pragma unroll
  for (int nt = 0; nt < 8; ++nt) b1v[nt] = b1[wv * 128 + nt * 16 + ln16];

#pragma unroll 1
  for (int t = 0; t < NSTEP; ++t) {
    __syncthreads();   // all waves done reading xi (GEMM1) and s1 (GEMM2)

    // ---- Poisson spike encoding: xi = (noise < x), f16 {0,1} in LDS --------
    const float* nz = noise + ((size_t)t * B_TOTAL + r0) * NIN;
    const float* xb = x + (size_t)r0 * NIN;
    for (int i = tid; i < BM * (NIN / 4); i += NTHREADS) {
      int r = i / (NIN / 4), c4 = i - r * (NIN / 4);
      v4f u  = __builtin_nontemporal_load((const v4f*)(nz + (size_t)r * NIN + 4 * c4));
      v4f xv = *(const v4f*)(xb + (size_t)r * NIN + 4 * c4);   // L2-resident
      _Float16* p = xi + r * XI_PITCH + 4 * c4;
      p[0] = (u[0] < xv[0]) ? (_Float16)1.0f : (_Float16)0.0f;
      p[1] = (u[1] < xv[1]) ? (_Float16)1.0f : (_Float16)0.0f;
      p[2] = (u[2] < xv[2]) ? (_Float16)1.0f : (_Float16)0.0f;
      p[3] = (u[3] < xv[3]) ? (_Float16)1.0f : (_Float16)0.0f;
    }
    __syncthreads();   // xi ready

    // ---- layer 1: accumulate i1 = xi @ W1^T on top of decay*v1*(1-s1) -----
#pragma unroll 1
    for (int k0 = 0; k0 < K1P; k0 += 32) {
      v16h a[4];
#pragma unroll
      for (int rt = 0; rt < 4; ++rt) {
        const _Float16* ar = xi + (rt * 16 + ln16) * XI_PITCH;
        a[rt] = cat16(*(const v8h*)(ar + k0 + hi * 8),
                      *(const v8h*)(ar + k0 + 16 + hi * 8));
      }
#pragma unroll
      for (int nt = 0; nt < 8; ++nt) {
        int n = wv * 128 + nt * 16 + ln16;
        const _Float16* wp = W1h + (size_t)n * K1P + k0 + hi * 8;
        v16h bmat = cat16(*(const v8h*)wp, *(const v8h*)(wp + 16));
#pragma unroll
        for (int rt = 0; rt < 4; ++rt)
          v1acc[rt][nt] = __builtin_amdgcn_wmma_f32_16x16x32_f16(
              false, a[rt], false, bmat, (short)0, v1acc[rt][nt], false, false);
      }
    }
    // bias, spike, publish s1, and pre-apply decay+reset for next step
#pragma unroll
    for (int rt = 0; rt < 4; ++rt)
#pragma unroll
      for (int nt = 0; nt < 8; ++nt) {
        int n = wv * 128 + nt * 16 + ln16;
#pragma unroll
        for (int j = 0; j < 8; ++j) {
          int m = rt * 16 + j + 8 * hi;
          float v = v1acc[rt][nt][j] + b1v[nt];
          float s = (v > 1.0f) ? 1.0f : 0.0f;
          s1[m * S1_PITCH + n] = (_Float16)s;
          v1acc[rt][nt][j] = DECAY1 * v * (1.0f - s);  // next step's C operand
        }
      }
    __syncthreads();   // s1 ready for all waves

    // ---- layer 2: accumulate i2 = s1 @ W2^T on top of decay*v2*(1-s2) -----
#pragma unroll 1
    for (int k0 = 0; k0 < NHID; k0 += 32) {
      v16h a[4];
#pragma unroll
      for (int rt = 0; rt < 4; ++rt) {
        const _Float16* ar = s1 + (rt * 16 + ln16) * S1_PITCH;
        a[rt] = cat16(*(const v8h*)(ar + k0 + hi * 8),
                      *(const v8h*)(ar + k0 + 16 + hi * 8));
      }
#pragma unroll
      for (int nt = 0; nt < 2; ++nt) {
        int n = wv * 32 + nt * 16 + ln16;
        const _Float16* wp = W2h + (size_t)n * NHID + k0 + hi * 8;
        v16h bmat = cat16(*(const v8h*)wp, *(const v8h*)(wp + 16));
#pragma unroll
        for (int rt = 0; rt < 4; ++rt)
          v2acc[rt][nt] = __builtin_amdgcn_wmma_f32_16x16x32_f16(
              false, a[rt], false, bmat, (short)0, v2acc[rt][nt], false, false);
      }
    }
#pragma unroll
    for (int rt = 0; rt < 4; ++rt)
#pragma unroll
      for (int nt = 0; nt < 2; ++nt)
#pragma unroll
        for (int j = 0; j < 8; ++j) {
          float v = v2acc[rt][nt][j];
          float s = (v > 1.0f) ? 1.0f : 0.0f;
          nsp[rt][nt][j] += s;
          v2acc[rt][nt][j] = DECAY2 * v * (1.0f - s);  // next step's C operand
        }
  }

  // ---- write spike counts [4096,256] f32 ------------------------------------
#pragma unroll
  for (int rt = 0; rt < 4; ++rt)
#pragma unroll
    for (int nt = 0; nt < 2; ++nt) {
      int n = wv * 32 + nt * 16 + ln16;
#pragma unroll
      for (int j = 0; j < 8; ++j) {
        int row = r0 + rt * 16 + j + 8 * hi;
        __builtin_nontemporal_store(nsp[rt][nt][j], out + (size_t)row * NOUT + n);
      }
    }
}

extern "C" void kernel_launch(void* const* d_in, const int* in_sizes, int n_in,
                              void* d_out, int out_size, void* d_ws, size_t ws_size,
                              hipStream_t stream) {
  (void)in_sizes; (void)n_in; (void)out_size; (void)ws_size;
  const float* x     = (const float*)d_in[0];
  const float* W1    = (const float*)d_in[1];
  const float* b1    = (const float*)d_in[2];
  const float* W2    = (const float*)d_in[3];
  const float* noise = (const float*)d_in[4];
  float* out = (float*)d_out;

  _Float16* W1h = (_Float16*)d_ws;                                   // 1024*800*2 B
  _Float16* W2h = (_Float16*)((char*)d_ws + (size_t)NHID * K1P * 2); // 256*1024*2 B

  cvt_w1<<<(NHID * K1P + 255) / 256, 256, 0, stream>>>(W1, W1h);
  cvt_w2<<<(NOUT * NHID + 255) / 256, 256, 0, stream>>>(W2, W2h);

  size_t smem = (size_t)BM * XI_PITCH * 2 + (size_t)BM * S1_PITCH * 2; // ~238 KB
  snn_kernel<<<B_TOTAL / BM, NTHREADS, smem, stream>>>(x, b1, W1h, W2h, noise, out);
}
